// SelfAttention_38122129719491
// MI455X (gfx1250) — matrix operand order repaired and checked by reading, not compiled, not run on hardware
//
#include <hip/hip_runtime.h>

typedef __attribute__((ext_vector_type(16))) __bf16 v16bf;
typedef __attribute__((ext_vector_type(8)))  float  v8f;

#define WMMA_BF16(A, B, C) \
  __builtin_amdgcn_wmma_f32_16x16x32_bf16(false, (A), false, (B), (short)0, (C), false, false)

static constexpr int CDIM  = 512;
static constexpr int NDIM  = 1024;   // H*W
static constexpr int BSZ   = 8;
static constexpr int HEADS = 8;
static constexpr int DHEAD = 64;

// ---------------------------------------------------------------------------
// f32 -> bf16 conversion (weights, vectorized x4)
// ---------------------------------------------------------------------------
__global__ void cvt_f32_bf16_x4(const float* __restrict__ s, __bf16* __restrict__ d, int n)
{
    int i = (blockIdx.x * 256 + threadIdx.x) * 4;
    if (i < n) {
        float4 v = *(const float4*)(s + i);
        d[i + 0] = (__bf16)v.x;
        d[i + 1] = (__bf16)v.y;
        d[i + 2] = (__bf16)v.z;
        d[i + 3] = (__bf16)v.w;
    }
}

// ---------------------------------------------------------------------------
// x [B,C,N] f32  ->  xt [B,N,C] bf16  (LDS 32x32 tile transpose)
// grid (N/32, C/32, B), block (32, 8)
// ---------------------------------------------------------------------------
__global__ __launch_bounds__(256)
void transpose_cvt(const float* __restrict__ x, __bf16* __restrict__ xt)
{
    __shared__ float tile[32][33];
    const int b  = blockIdx.z;
    const int n0 = blockIdx.x * 32;
    const int c0 = blockIdx.y * 32;
    const float* xs = x  + (size_t)b * CDIM * NDIM;
    __bf16*      xd = xt + (size_t)b * NDIM * CDIM;
#pragma unroll
    for (int i = 0; i < 32; i += 8) {
        int c = c0 + threadIdx.y + i;
        tile[threadIdx.y + i][threadIdx.x] = xs[(size_t)c * NDIM + n0 + threadIdx.x];
    }
    __syncthreads();
#pragma unroll
    for (int i = 0; i < 32; i += 8) {
        int n = n0 + threadIdx.y + i;
        xd[(size_t)n * CDIM + c0 + threadIdx.x] = (__bf16)tile[threadIdx.x][threadIdx.y + i];
    }
}

// ---------------------------------------------------------------------------
// Shared GEMM tile: Y[64o x 16n] = W[512x512] * X[512x1024], bf16 WMMA, f32 acc.
// Xt is X transposed [N x C] so the B fragment is one contiguous 16-element run
// per lane (k = 16*half .. +15), and A fragments are two contiguous 8-runs.
// ---------------------------------------------------------------------------
__device__ inline void gemm_tile_64x16(const __bf16* __restrict__ W,
                                       const __bf16* __restrict__ Xt,
                                       int o0, int n0, v8f acc[4])
{
    const int lane = threadIdx.x & 31;
    const int ncol = lane & 15;
    const int half = lane >> 4;
    const __bf16* xrow = Xt + (size_t)(n0 + ncol) * CDIM;

    for (int k0 = 0; k0 < CDIM; k0 += 32) {
        v16bf Bf;
#pragma unroll
        for (int e = 0; e < 16; ++e) Bf[e] = xrow[k0 + 16 * half + e];   // 2x b128
#pragma unroll
        for (int t = 0; t < 4; ++t) {
            const __bf16* wrow = W + (size_t)(o0 + t * 16 + ncol) * CDIM;
            v16bf Af;
#pragma unroll
            for (int e = 0; e < 8; ++e) Af[e]     = wrow[k0 + 16 * half + e];      // b128
#pragma unroll
            for (int e = 0; e < 8; ++e) Af[8 + e] = wrow[k0 + 16 * half + 8 + e];  // b128
            acc[t] = WMMA_BF16(Af, Bf, acc[t]);
        }
    }
}

// ---------------------------------------------------------------------------
// QKV projection. Q,K stored TRANSPOSED per (b,h): [n][c_local] for fast
// attention fragment loads; V stored [c][n] (its attention A-fragment strides
// along keys). grid (16, 8, B*3), block 128 (4 waves).
// ---------------------------------------------------------------------------
__global__ __launch_bounds__(128)
void qkv_kernel(const __bf16* __restrict__ xt,
                const __bf16* __restrict__ wq, const __bf16* __restrict__ wk,
                const __bf16* __restrict__ wv,
                const float* __restrict__ bq, const float* __restrict__ bk,
                const float* __restrict__ bv,
                __bf16* __restrict__ Qt, __bf16* __restrict__ Kt, __bf16* __restrict__ V)
{
    const int wave = threadIdx.x >> 5;
    const int lane = threadIdx.x & 31;
    const int ncol = lane & 15;
    const int half = lane >> 4;
    const int n0 = (blockIdx.x * 4 + wave) * 16;
    const int o0 = blockIdx.y * 64;
    const int b  = blockIdx.z / 3;
    const int m  = blockIdx.z % 3;

    const __bf16* W    = (m == 0) ? wq : (m == 1) ? wk : wv;
    const float*  bias = (m == 0) ? bq : (m == 1) ? bk : bv;

    const __bf16* X = xt + (size_t)b * NDIM * CDIM;

    const v8f z8 = {0.f, 0.f, 0.f, 0.f, 0.f, 0.f, 0.f, 0.f};
    v8f acc[4] = {z8, z8, z8, z8};
    gemm_tile_64x16(W, X, o0, n0, acc);

    if (m == 2) {   // V: [b][c][n]
        __bf16* out = V + (size_t)b * CDIM * NDIM;
#pragma unroll
        for (int t = 0; t < 4; ++t)
#pragma unroll
            for (int r = 0; r < 8; ++r) {
                int orow = o0 + t * 16 + r + 8 * half;       // D layout: M = r+8*half
                out[(size_t)orow * NDIM + n0 + ncol] = (__bf16)(acc[t][r] + bias[orow]);
            }
    } else {        // Q/K: [b][h][n][c_local]
        __bf16* out = (m == 0) ? Qt : Kt;
#pragma unroll
        for (int t = 0; t < 4; ++t)
#pragma unroll
            for (int r = 0; r < 8; ++r) {
                int orow = o0 + t * 16 + r + 8 * half;
                int hh = orow >> 6, cl = orow & 63;
                out[((size_t)(b * HEADS + hh) * NDIM + n0 + ncol) * DHEAD + cl] =
                    (__bf16)(acc[t][r] + bias[orow]);
            }
    }
}

// ---------------------------------------------------------------------------
// Flash attention per wave: 16 queries of one (b,h). S^T = K^T Q so softmax
// runs along D-tile rows (in-lane + one shfl_xor(16)). Online max/sum, fused
// "+ value(x)" residual, Z written TRANSPOSED [b][n][c] for the projection.
// grid (16, heads, B), block 128.
// ---------------------------------------------------------------------------
__global__ __launch_bounds__(128)
void attn_kernel(const __bf16* __restrict__ Qt, const __bf16* __restrict__ Kt,
                 const __bf16* __restrict__ V, __bf16* __restrict__ Zt)
{
    const int lane = threadIdx.x & 31;
    const int wave = threadIdx.x >> 5;
    const int ncol = lane & 15;
    const int half = lane >> 4;
    const int b = blockIdx.z, h = blockIdx.y;
    const int n0 = (blockIdx.x * 4 + wave) * 16;

    const __bf16* Qh = Qt + (size_t)(b * HEADS + h) * NDIM * DHEAD;
    const __bf16* Kh = Kt + (size_t)(b * HEADS + h) * NDIM * DHEAD;
    const __bf16* Vh = V  + ((size_t)b * CDIM + (size_t)h * DHEAD) * NDIM;

    // Q held for the whole loop: B fragments, two b128 each (contiguous c-run).
    const __bf16* qrow = Qh + (size_t)(n0 + ncol) * DHEAD;
    v16bf Bq[2];
#pragma unroll
    for (int f = 0; f < 2; ++f)
#pragma unroll
        for (int e = 0; e < 16; ++e) Bq[f][e] = qrow[32 * f + 16 * half + e];

    const v8f z8 = {0.f, 0.f, 0.f, 0.f, 0.f, 0.f, 0.f, 0.f};
    v8f Oacc[4] = {z8, z8, z8, z8};
    float run_max = -3.0e38f, run_sum = 0.f;

    for (int m0 = 0; m0 < NDIM; m0 += 32) {
        if (m0 + 32 < NDIM) __builtin_prefetch(Kh + (size_t)(m0 + 32) * DHEAD, 0, 0);

        // --- S^T tiles (16m x 16n), accumulate over 2 c-chunks ---
        v8f ST[2];
#pragma unroll
        for (int s = 0; s < 2; ++s) {
            v8f c0 = z8;
            const __bf16* krow = Kh + (size_t)(m0 + 16 * s + ncol) * DHEAD;
#pragma unroll
            for (int f = 0; f < 2; ++f) {
                v16bf Ak;
#pragma unroll
                for (int e = 0; e < 8; ++e) Ak[e]     = krow[32 * f + 16 * half + e];
#pragma unroll
                for (int e = 0; e < 8; ++e) Ak[8 + e] = krow[32 * f + 16 * half + 8 + e];
                c0 = WMMA_BF16(Ak, Bq[f], c0);
            }
            ST[s] = c0;
        }

        // --- online softmax over keys (scale = 1/sqrt(64) = 0.125) ---
        float tmax = -3.0e38f;
#pragma unroll
        for (int s = 0; s < 2; ++s)
#pragma unroll
            for (int r = 0; r < 8; ++r) {
                float v = ST[s][r] * 0.125f;
                ST[s][r] = v;
                tmax = fmaxf(tmax, v);
            }
        tmax = fmaxf(tmax, __shfl_xor(tmax, 16, 32));
        float nmax = fmaxf(run_max, tmax);
        float corr = __expf(run_max - nmax);
        float tsum = 0.f;
#pragma unroll
        for (int s = 0; s < 2; ++s)
#pragma unroll
            for (int r = 0; r < 8; ++r) {
                float p = __expf(ST[s][r] - nmax);
                ST[s][r] = p;
                tsum += p;
            }
        tsum += __shfl_xor(tsum, 16, 32);
        run_sum = run_sum * corr + tsum;
        run_max = nmax;
#pragma unroll
        for (int t = 0; t < 4; ++t)
#pragma unroll
            for (int r = 0; r < 8; ++r) Oacc[t][r] *= corr;

        // --- repack P^T (D layout) into a WMMA B fragment (k = local m) ---
        float A0[8], A1[8];   // tile s=half values held at half-0 / half-1 lanes
#pragma unroll
        for (int r = 0; r < 8; ++r) {
            float own0 = ST[0][r], own1 = ST[1][r];
            float oth0 = __shfl_xor(own0, 16, 32);
            float oth1 = __shfl_xor(own1, 16, 32);
            float selo = half ? own1 : own0;
            float selp = half ? oth1 : oth0;
            A0[r] = half ? selp : selo;
            A1[r] = half ? selo : selp;
        }
        v16bf Bp;
#pragma unroll
        for (int j = 0; j < 4; ++j) {
            Bp[2 * j]     = (__bf16)A0[2 * j];
            Bp[2 * j + 1] = (__bf16)A0[2 * j + 1];
        }
#pragma unroll
        for (int j = 4; j < 8; ++j) {
            Bp[2 * j]     = (__bf16)A1[2 * j - 8];
            Bp[2 * j + 1] = (__bf16)A1[2 * j - 7];
        }

        // --- O += V_tile * P^T (4 c-groups of 16 rows, K = 32 keys) ---
#pragma unroll
        for (int cg = 0; cg < 4; ++cg) {
            const __bf16* vrow = Vh + (size_t)(cg * 16 + ncol) * NDIM;
            v16bf Av;
#pragma unroll
            for (int e = 0; e < 8; ++e) Av[e]     = vrow[m0 + 16 * half + e];
#pragma unroll
            for (int e = 0; e < 8; ++e) Av[8 + e] = vrow[m0 + 16 * half + 8 + e];
            Oacc[cg] = WMMA_BF16(Av, Bp, Oacc[cg]);
        }
    }

    // --- epilogue: normalize, fuse residual (+ value(x)), store Z transposed ---
    float inv = 1.0f / run_sum;
    __bf16* zrow = Zt + ((size_t)b * NDIM + n0 + ncol) * CDIM + (size_t)h * DHEAD;
#pragma unroll
    for (int cg = 0; cg < 4; ++cg)
#pragma unroll
        for (int r = 0; r < 8; ++r) {
            int cl = cg * 16 + r + 8 * half;                 // D layout M = r+8*half
            float o = Oacc[cg][r] * inv + (float)Vh[(size_t)cl * NDIM + n0 + ncol];
            zrow[cl] = (__bf16)o;
        }
}

// ---------------------------------------------------------------------------
// Output projection: out = wo * Z + bo (Z consumed transposed), f32 store.
// grid (16, 8, B), block 128.
// ---------------------------------------------------------------------------
__global__ __launch_bounds__(128)
void proj_kernel(const __bf16* __restrict__ Zt, const __bf16* __restrict__ wo,
                 const float* __restrict__ bo, float* __restrict__ out)
{
    const int wave = threadIdx.x >> 5;
    const int lane = threadIdx.x & 31;
    const int ncol = lane & 15;
    const int half = lane >> 4;
    const int n0 = (blockIdx.x * 4 + wave) * 16;
    const int o0 = blockIdx.y * 64;
    const int b  = blockIdx.z;

    const __bf16* X = Zt + (size_t)b * NDIM * CDIM;
    float* Y = out + (size_t)b * CDIM * NDIM;

    const v8f z8 = {0.f, 0.f, 0.f, 0.f, 0.f, 0.f, 0.f, 0.f};
    v8f acc[4] = {z8, z8, z8, z8};
    gemm_tile_64x16(wo, X, o0, n0, acc);

#pragma unroll
    for (int t = 0; t < 4; ++t)
#pragma unroll
        for (int r = 0; r < 8; ++r) {
            int orow = o0 + t * 16 + r + 8 * half;
            Y[(size_t)orow * NDIM + n0 + ncol] = acc[t][r] + bo[orow];
        }
}

// ---------------------------------------------------------------------------
extern "C" void kernel_launch(void* const* d_in, const int* in_sizes, int n_in,
                              void* d_out, int out_size, void* d_ws, size_t ws_size,
                              hipStream_t stream)
{
    const float* x  = (const float*)d_in[0];
    const float* wq = (const float*)d_in[1];
    const float* bq = (const float*)d_in[2];
    const float* wk = (const float*)d_in[3];
    const float* bk = (const float*)d_in[4];
    const float* wv = (const float*)d_in[5];
    const float* bv = (const float*)d_in[6];
    const float* wo = (const float*)d_in[7];
    const float* bo = (const float*)d_in[8];
    // d_in[9] = heads (==8), baked into constants.

    const size_t ACT = (size_t)BSZ * CDIM * NDIM;   // 4,194,304 elements
    const size_t WEL = (size_t)CDIM * CDIM;         //   262,144 elements

    __bf16* ws  = (__bf16*)d_ws;
    __bf16* xt  = ws;  ws += ACT;   // x transposed  [B][N][C]
    __bf16* wqb = ws;  ws += WEL;
    __bf16* wkb = ws;  ws += WEL;
    __bf16* wvb = ws;  ws += WEL;
    __bf16* wob = ws;  ws += WEL;
    __bf16* Qt  = ws;  ws += ACT;   // [B][H][N][64]
    __bf16* Kt  = ws;  ws += ACT;   // [B][H][N][64]
    __bf16* V   = ws;  ws += ACT;   // [B][C][N]
    __bf16* Zt  = ws;  ws += ACT;   // [B][N][C]

    transpose_cvt<<<dim3(NDIM / 32, CDIM / 32, BSZ), dim3(32, 8), 0, stream>>>(x, xt);
    cvt_f32_bf16_x4<<<(int)(WEL / 4 / 256), 256, 0, stream>>>(wq, wqb, (int)WEL);
    cvt_f32_bf16_x4<<<(int)(WEL / 4 / 256), 256, 0, stream>>>(wk, wkb, (int)WEL);
    cvt_f32_bf16_x4<<<(int)(WEL / 4 / 256), 256, 0, stream>>>(wv, wvb, (int)WEL);
    cvt_f32_bf16_x4<<<(int)(WEL / 4 / 256), 256, 0, stream>>>(wo, wob, (int)WEL);

    qkv_kernel<<<dim3(16, 8, BSZ * 3), 128, 0, stream>>>(
        xt, wqb, wkb, wvb, bq, bk, bv, Qt, Kt, V);

    attn_kernel<<<dim3(16, HEADS, BSZ), 128, 0, stream>>>(Qt, Kt, V, Zt);

    proj_kernel<<<dim3(16, 8, BSZ), 128, 0, stream>>>(Zt, wob, bo, (float*)d_out);
}
